// AugmentedMemoryTransformerEncoderLayer_16200616641182
// MI455X (gfx1250) — compile-verified, matrix operand order checked by reading, not run on hardware
//
#include <hip/hip_runtime.h>
#include <hip/hip_bf16.h>

// ---------------------------------------------------------------------------
// AugmentedMemoryTransformerEncoderLayer for MI455X (gfx1250, wave32, WMMA)
// ---------------------------------------------------------------------------
constexpr int kL  = 384;
constexpr int kB  = 16;
constexpr int kD  = 1024;
constexpr int kH  = 16;
constexpr int kM  = 64;     // memory length
constexpr int kF  = 4096;
constexpr int kHD = kD / kH;        // 64
constexpr int kNKEY = kM + kL;      // 448 keys per head
constexpr float kLNEPS = 1e-5f;
constexpr float kSTDSCALE = 0.5f;

typedef __bf16 bf16_t;
typedef __attribute__((ext_vector_type(8)))  __bf16 v8bf;
typedef __attribute__((ext_vector_type(16))) __bf16 v16bf;
typedef __attribute__((ext_vector_type(8)))  float  v8f;

union FragU { v16bf v; v8bf h2[2]; };

// ---------------- wave32 reductions -----------------
__device__ __forceinline__ float wave_sum(float v) {
    #pragma unroll
    for (int o = 16; o; o >>= 1) v += __shfl_xor(v, o, 32);
    return v;
}
__device__ __forceinline__ float wave_max(float v) {
    #pragma unroll
    for (int o = 16; o; o >>= 1) v = fmaxf(v, __shfl_xor(v, o, 32));
    return v;
}

// ---------------- f32 -> bf16 convert -----------------
__global__ void cvt_bf16_kernel(const float* __restrict__ src,
                                bf16_t* __restrict__ dst, int n) {
    for (int i = blockIdx.x * blockDim.x + threadIdx.x; i < n;
         i += gridDim.x * blockDim.x)
        dst[i] = (bf16_t)src[i];
}

// ---------------- LayerNorm (one row of D=1024 per block) -----------------
__global__ __launch_bounds__(256) void ln_kernel(const float* __restrict__ X,
                                                 const float* __restrict__ g,
                                                 const float* __restrict__ be,
                                                 bf16_t* __restrict__ o0,
                                                 bf16_t* __restrict__ o1) {
    const int row = blockIdx.x;
    const float* xr = X + (size_t)row * kD;
    float s = 0.f, s2 = 0.f;
    for (int i = threadIdx.x; i < kD; i += 256) {
        float v = xr[i]; s += v; s2 += v * v;
    }
    s = wave_sum(s); s2 = wave_sum(s2);
    __shared__ float ls[8], ls2[8];
    __shared__ float smu, srs;
    const int wid = threadIdx.x >> 5;
    if ((threadIdx.x & 31) == 0) { ls[wid] = s; ls2[wid] = s2; }
    __syncthreads();
    if (threadIdx.x == 0) {
        float ts = 0.f, ts2 = 0.f;
        #pragma unroll
        for (int i = 0; i < 8; ++i) { ts += ls[i]; ts2 += ls2[i]; }
        float mu  = ts / kD;
        float var = ts2 / kD - mu * mu;
        smu = mu; srs = rsqrtf(var + kLNEPS);
    }
    __syncthreads();
    const float mu = smu, rs = srs;
    for (int i = threadIdx.x; i < kD; i += 256) {
        float v = (xr[i] - mu) * rs * g[i] + be[i];
        bf16_t bv = (bf16_t)v;
        o0[(size_t)row * kD + i] = bv;
        if (o1) o1[(size_t)row * kD + i] = bv;
    }
}

// ---------------- summary = mean over tokens [64,320) -----------------
__global__ void summary_kernel(bf16_t* __restrict__ inp_bf) {
    int idx = blockIdx.x * blockDim.x + threadIdx.x;  // over B*D
    if (idx >= kB * kD) return;
    const int b = idx / kD, d = idx % kD;
    float s = 0.f;
    for (int l = kM; l < kL - kM; ++l)  // LEFT=64 .. L-RIGHT=320
        s += (float)inp_bf[((size_t)l * kB + b) * kD + d];
    inp_bf[((size_t)kL * kB + b) * kD + d] = (bf16_t)(s * (1.f / 256.f));
}

// ---------------- register-blocked bf16 NT GEMM ----------------------------
// One wave computes a 32x64 C tile: 2 M-frags x 4 N-frags = 8 WMMAs / k-step,
// software-pipelined over k-steps of 32 (K % 64 == 0 for all layers here).
// C[m,n] = act(alpha * (sum_k A[m,k]*W[n,k] + bias[n])) (+ resid[m,n])
__device__ __forceinline__ void load_a_frags(FragU (&A)[2],
                                             const bf16_t* __restrict__ Abase,
                                             int Mrows, int tm, int K, int kk) {
    const int lane = threadIdx.x & 31;
    const int r    = lane & 15;
    const int kb   = (lane >> 4) << 3;
    #pragma unroll
    for (int mi = 0; mi < 2; ++mi) {
        int row = tm * 32 + mi * 16 + r;
        if (row >= Mrows) row = Mrows - 1;          // clamp (stores guarded)
        const bf16_t* p = Abase + (size_t)row * K + kk + kb;
        A[mi].h2[0] = *(const v8bf*)p;
        A[mi].h2[1] = *(const v8bf*)(p + 16);
    }
}
__device__ __forceinline__ void load_b_frags(FragU (&Bv)[4],
                                             const bf16_t* __restrict__ Wbase,
                                             int tn, int K, int kk) {
    const int lane = threadIdx.x & 31;
    const int r    = lane & 15;
    const int kb   = (lane >> 4) << 3;
    #pragma unroll
    for (int ni = 0; ni < 4; ++ni) {
        const bf16_t* p = Wbase + (size_t)(tn * 64 + ni * 16 + r) * K + kk + kb;
        Bv[ni].h2[0] = *(const v8bf*)p;
        Bv[ni].h2[1] = *(const v8bf*)(p + 16);
    }
}
__device__ __forceinline__ void mma_step(v8f (&acc)[8], FragU (&A)[2], FragU (&Bv)[4]) {
    #pragma unroll
    for (int mi = 0; mi < 2; ++mi)
        #pragma unroll
        for (int ni = 0; ni < 4; ++ni)
            acc[mi * 4 + ni] = __builtin_amdgcn_wmma_f32_16x16x32_bf16(
                false, A[mi].v, false, Bv[ni].v, (short)0, acc[mi * 4 + ni],
                false, false);
}

__global__ __launch_bounds__(32) void gemm_bf16_nt(
    const bf16_t* __restrict__ A, const bf16_t* __restrict__ W,
    const float* __restrict__ bias, const float* __restrict__ resid,
    float* __restrict__ outF, bf16_t* __restrict__ outB,
    int Mrows, int N, int K, float alpha, int relu) {
    const int tm = blockIdx.x, tn = blockIdx.y;
    const int lane = threadIdx.x & 31;

    v8f acc[8];
    #pragma unroll
    for (int i = 0; i < 8; ++i) acc[i] = v8f{};

    FragU A0[2], B0[4], A1[2], B1[4];
    load_a_frags(A0, A, Mrows, tm, K, 0);
    load_b_frags(B0, W, tn, K, 0);
    for (int k0 = 0; k0 < K; k0 += 64) {
        if (k0 + 192 < K) {   // prefetch two pipeline stages ahead
            int r = lane & 15;
            int arow = tm * 32 + r;      if (arow >= Mrows) arow = Mrows - 1;
            __builtin_prefetch(A + (size_t)arow * K + k0 + 192, 0, 1);
            __builtin_prefetch(W + (size_t)(tn * 64 + r) * K + k0 + 192, 0, 1);
        }
        if (k0 + 32 < K) {
            load_a_frags(A1, A, Mrows, tm, K, k0 + 32);
            load_b_frags(B1, W, tn, K, k0 + 32);
        }
        mma_step(acc, A0, B0);
        if (k0 + 64 < K) {
            load_a_frags(A0, A, Mrows, tm, K, k0 + 64);
            load_b_frags(B0, W, tn, K, k0 + 64);
        }
        mma_step(acc, A1, B1);
    }

    const int col   = lane & 15;
    const int rbase = (lane < 16) ? 0 : 8;
    #pragma unroll
    for (int ni = 0; ni < 4; ++ni) {
        const int n = tn * 64 + ni * 16 + col;
        const float bsv = bias ? bias[n] : 0.f;
        #pragma unroll
        for (int mi = 0; mi < 2; ++mi) {
            #pragma unroll
            for (int i = 0; i < 8; ++i) {
                int row = tm * 32 + mi * 16 + i + rbase;
                if (row >= Mrows) continue;
                float vv = alpha * (acc[mi * 4 + ni][i] + bsv);
                if (relu) vv = fmaxf(vv, 0.f);
                size_t idx = (size_t)row * N + n;
                if (resid) vv += resid[idx];
                if (outF) outF[idx] = vv;
                if (outB) outB[idx] = (bf16_t)vv;
            }
        }
    }
}

// ---------------- fused attention: one wave per (head, 16-query tile) ------
// scores (WMMA q.kT) -> softmax -> suppression -> softmax -> WMMA p.v
// v-tiles are staged into LDS with GLOBAL_LOAD_ASYNC_TO_LDS_B128 (ASYNCcnt).
__global__ __launch_bounds__(32) void attn_kernel(
    const bf16_t* __restrict__ q, const bf16_t* __restrict__ k,
    const bf16_t* __restrict__ v, bf16_t* __restrict__ o) {
    __shared__ float  s_sc[16][kNKEY];
    __shared__ __align__(16) bf16_t s_p[16][kNKEY];
    __shared__ __align__(16) bf16_t s_v[32][kHD];

    const int tile = blockIdx.x;          // 0..24 (ceil(385/16))
    const int n    = blockIdx.y;          // b*H + h
    const int b = n / kH, h = n % kH;
    const int lane  = threadIdx.x & 31;
    const int r     = lane & 15;
    const int kb    = (lane >> 4) << 3;
    const int rbase = (lane < 16) ? 0 : 8;
    const int col   = lane & 15;

    // ---- q fragments (2 K-steps of 32 over HD=64); clamp rows past 384 ----
    int t = tile * 16 + r; if (t > kL) t = kL;
    const bf16_t* qp = q + ((size_t)t * kB + b) * kD + h * kHD + kb;
    FragU a0, a1;
    a0.h2[0] = *(const v8bf*)(qp);      a0.h2[1] = *(const v8bf*)(qp + 16);
    a1.h2[0] = *(const v8bf*)(qp + 32); a1.h2[1] = *(const v8bf*)(qp + 48);

    // ---- phase 1: 16 x 448 scores via WMMA ----
    for (int kt = 0; kt < kNKEY / 16; ++kt) {
        const bf16_t* kp = k + ((size_t)(kt * 16 + r) * kB + b) * kD + h * kHD + kb;
        FragU b0, b1;
        b0.h2[0] = *(const v8bf*)(kp);      b0.h2[1] = *(const v8bf*)(kp + 16);
        b1.h2[0] = *(const v8bf*)(kp + 32); b1.h2[1] = *(const v8bf*)(kp + 48);
        v8f acc = {};
        acc = __builtin_amdgcn_wmma_f32_16x16x32_bf16(false, a0.v, false, b0.v,
                                                      (short)0, acc, false, false);
        acc = __builtin_amdgcn_wmma_f32_16x16x32_bf16(false, a1.v, false, b1.v,
                                                      (short)0, acc, false, false);
        #pragma unroll
        for (int i = 0; i < 8; ++i) s_sc[i + rbase][kt * 16 + col] = acc[i];
    }
    __syncthreads();

    // ---- phase 2: softmax -> suppression -> softmax, per query row ----
    for (int rr = 0; rr < 16; ++rr) {
        const int qi = tile * 16 + rr;
        if (qi > kL) break;                 // uniform across the wave
        float sl[14], pl[14];
        float m = -__builtin_inff();
        #pragma unroll
        for (int j = 0; j < 14; ++j) {
            int kk = lane + 32 * j;
            float s = s_sc[rr][kk];
            if (qi == kL && kk < kM) s = -__builtin_inff();  // mask mem keys of summary row
            sl[j] = s; m = fmaxf(m, s);
        }
        m = wave_max(m);
        float sum = 0.f;
        #pragma unroll
        for (int j = 0; j < 14; ++j) { pl[j] = __expf(sl[j] - m); sum += pl[j]; }
        sum = wave_sum(sum);
        const float inv = 1.f / sum;
        float nz = 0.f, sp = 0.f;
        #pragma unroll
        for (int j = 0; j < 14; ++j) {
            pl[j] *= inv;
            if (pl[j] > 0.f) { nz += 1.f; sp += pl[j]; }
        }
        nz = wave_sum(nz); sp = wave_sum(sp);
        const float mean = sp / nz;
        float dv = 0.f;
        #pragma unroll
        for (int j = 0; j < 14; ++j)
            if (pl[j] > 0.f) { float d = pl[j] - mean; dv += d * d; }
        dv = wave_sum(dv);
        const float thr = mean - kSTDSCALE * sqrtf(dv / (nz - 1.f));
        float m2 = -__builtin_inff();
        #pragma unroll
        for (int j = 0; j < 14; ++j) {
            float s2v = (pl[j] < thr) ? -__builtin_inff() : sl[j];
            sl[j] = s2v; m2 = fmaxf(m2, s2v);
        }
        m2 = wave_max(m2);
        float sum2 = 0.f;
        #pragma unroll
        for (int j = 0; j < 14; ++j) { float e = __expf(sl[j] - m2); sl[j] = e; sum2 += e; }
        sum2 = wave_sum(sum2);
        const float inv2 = 1.f / sum2;
        #pragma unroll
        for (int j = 0; j < 14; ++j)
            s_p[rr][lane + 32 * j] = (bf16_t)(sl[j] * inv2);
    }
    __syncthreads();

    // ---- phase 3: attn = p @ v via WMMA; v tiles staged memory->LDS with
    //      GLOBAL_LOAD_ASYNC_TO_LDS_B128 (no VGPR round-trip, ASYNCcnt) ----
    v8f on[4] = {v8f{}, v8f{}, v8f{}, v8f{}};
    for (int kt2 = 0; kt2 < kNKEY / 32; ++kt2) {
        {   // async stage: 32 keys x 64 bf16 (each lane copies one key row)
            const bf16_t* vp = v + ((size_t)(kt2 * 32 + lane) * kB + b) * kD + h * kHD;
            unsigned lds_base = (unsigned)(uintptr_t)&s_v[lane][0];
            unsigned long long ga = (unsigned long long)(uintptr_t)vp;
            #pragma unroll
            for (int c = 0; c < 8; ++c) {
                unsigned lds_addr = lds_base + c * 16;
                unsigned long long gaddr = ga + c * 16;
                asm volatile("global_load_async_to_lds_b128 %0, %1, off"
                             :: "v"(lds_addr), "v"(gaddr)
                             : "memory");
            }
            asm volatile("s_wait_asynccnt 0x0" ::: "memory");
        }
        __syncthreads();
        FragU a;
        const bf16_t* pp = &s_p[r][kt2 * 32 + kb];
        a.h2[0] = *(const v8bf*)pp; a.h2[1] = *(const v8bf*)(pp + 16);
        #pragma unroll
        for (int nt = 0; nt < 4; ++nt) {
            FragU bb;
            #pragma unroll
            for (int j = 0; j < 16; ++j) {
                int kk = kb + j + ((j >= 8) ? 8 : 0);
                bb.v[j] = s_v[kk][nt * 16 + col];
            }
            on[nt] = __builtin_amdgcn_wmma_f32_16x16x32_bf16(false, a.v, false, bb.v,
                                                             (short)0, on[nt], false, false);
        }
        __syncthreads();
    }
    #pragma unroll
    for (int nt = 0; nt < 4; ++nt)
        #pragma unroll
        for (int i = 0; i < 8; ++i) {
            int qrow = tile * 16 + i + rbase;
            if (qrow <= kL)
                o[((size_t)qrow * kB + b) * kD + h * kHD + nt * 16 + col] =
                    (bf16_t)on[nt][i];
        }
}

// ---------------------------------------------------------------------------
extern "C" void kernel_launch(void* const* d_in, const int* in_sizes, int n_in,
                              void* d_out, int out_size, void* d_ws, size_t ws_size,
                              hipStream_t stream) {
    const float* x      = (const float*)d_in[0];
    const float* memory = (const float*)d_in[1];
    const float* q_w  = (const float*)d_in[2];
    const float* q_b  = (const float*)d_in[3];
    const float* k_w  = (const float*)d_in[4];
    const float* k_b  = (const float*)d_in[5];
    const float* v_w  = (const float*)d_in[6];
    const float* v_b  = (const float*)d_in[7];
    const float* out_w = (const float*)d_in[8];
    const float* out_b = (const float*)d_in[9];
    const float* ln1_w = (const float*)d_in[10];
    const float* ln1_b = (const float*)d_in[11];
    const float* fc1_w = (const float*)d_in[12];
    const float* fc1_b = (const float*)d_in[13];
    const float* fc2_w = (const float*)d_in[14];
    const float* fc2_b = (const float*)d_in[15];
    const float* ln2_w = (const float*)d_in[16];
    const float* ln2_b = (const float*)d_in[17];

    // ---- carve workspace ----
    char* wsp = (char*)d_ws;
    auto alloc = [&](size_t bytes) -> void* {
        void* p = (void*)wsp;
        wsp += (bytes + 255) & ~(size_t)255;
        return p;
    };
    const size_t DD = (size_t)kD * kD;            // 1M
    const size_t FD = (size_t)kF * kD;            // 4M
    const size_t NQ  = (size_t)(kL + 1) * kB;     // 6160 query rows
    const size_t NKV = (size_t)kNKEY * kB;        // 7168 kv rows
    const size_t NX  = (size_t)kL * kB;           // 6144 token rows

    bf16_t* wq   = (bf16_t*)alloc(DD * 2);
    bf16_t* wk   = (bf16_t*)alloc(DD * 2);
    bf16_t* wv   = (bf16_t*)alloc(DD * 2);
    bf16_t* wo   = (bf16_t*)alloc(DD * 2);
    bf16_t* wfc1 = (bf16_t*)alloc(FD * 2);
    bf16_t* wfc2 = (bf16_t*)alloc(FD * 2);
    bf16_t* inp_bf  = (bf16_t*)alloc(NQ  * kD * 2);   // [xn ; summary]
    bf16_t* kvin_bf = (bf16_t*)alloc(NKV * kD * 2);   // [memory ; xn]
    bf16_t* q_bf    = (bf16_t*)alloc(NQ  * kD * 2);
    bf16_t* k_bf    = (bf16_t*)alloc(NKV * kD * 2);
    bf16_t* v_bf    = (bf16_t*)alloc(NKV * kD * 2);
    bf16_t* attn_bf = (bf16_t*)alloc(NQ  * kD * 2);
    bf16_t* xn2_bf  = (bf16_t*)alloc(NX * kD * 2);
    bf16_t* h1_bf   = (bf16_t*)alloc(NX * kF * 2);
    float*  x2      = (float*) alloc(NX * kD * 4);
    (void)ws_size; (void)n_in; (void)in_sizes; (void)out_size;

    auto cvt = [&](const float* s, bf16_t* d, size_t n) {
        int blocks = (int)((n + 255) / 256);
        cvt_bf16_kernel<<<blocks, 256, 0, stream>>>(s, d, (int)n);
    };
    // weight + memory converts (fp32 -> bf16)
    cvt(q_w, wq, DD); cvt(k_w, wk, DD); cvt(v_w, wv, DD); cvt(out_w, wo, DD);
    cvt(fc1_w, wfc1, FD); cvt(fc2_w, wfc2, FD);
    cvt(memory, kvin_bf, (size_t)kM * kB * kD);       // kv rows [0, M*B)

    // LN1: writes xn into inp rows [0, L*B) and kv rows [M*B, ...)
    ln_kernel<<<(int)NX, 256, 0, stream>>>(x, ln1_w, ln1_b, inp_bf,
                                           kvin_bf + (size_t)kM * kB * kD);
    // summary token -> inp rows [L*B, (L+1)*B)
    summary_kernel<<<(kB * kD + 255) / 256, 256, 0, stream>>>(inp_bf);

    const float scaling = 0.125f;   // HD^-0.5 = 64^-0.5
    // q/k/v projections (32x64 wave tiles)
    gemm_bf16_nt<<<dim3((int)((NQ + 31) / 32), kD / 64), 32, 0, stream>>>(
        inp_bf, wq, q_b, nullptr, nullptr, q_bf, (int)NQ, kD, kD, scaling, 0);
    gemm_bf16_nt<<<dim3((int)((NKV + 31) / 32), kD / 64), 32, 0, stream>>>(
        kvin_bf, wk, k_b, nullptr, nullptr, k_bf, (int)NKV, kD, kD, 1.f, 0);
    gemm_bf16_nt<<<dim3((int)((NKV + 31) / 32), kD / 64), 32, 0, stream>>>(
        kvin_bf, wv, v_b, nullptr, nullptr, v_bf, (int)NKV, kD, kD, 1.f, 0);

    // fused attention (25 query tiles x 256 head-batches)
    attn_kernel<<<dim3((kL + 1 + 15) / 16, kB * kH), 32, 0, stream>>>(
        q_bf, k_bf, v_bf, attn_bf);

    // out-proj over the kept L*B rows, fused residual-add: x2 = x + attn@Wo + b
    gemm_bf16_nt<<<dim3((int)((NX + 31) / 32), kD / 64), 32, 0, stream>>>(
        attn_bf, wo, out_b, x, x2, nullptr, (int)NX, kD, kD, 1.f, 0);

    // LN2
    ln_kernel<<<(int)NX, 256, 0, stream>>>(x2, ln2_w, ln2_b, xn2_bf, nullptr);

    // FFN: h1 = relu(xn2 @ Wfc1^T + b)   ;   out = x2 + h1 @ Wfc2^T + b
    gemm_bf16_nt<<<dim3((int)((NX + 31) / 32), kF / 64), 32, 0, stream>>>(
        xn2_bf, wfc1, fc1_b, nullptr, nullptr, h1_bf, (int)NX, kF, kD, 1.f, 1);
    gemm_bf16_nt<<<dim3((int)((NX + 31) / 32), kD / 64), 32, 0, stream>>>(
        h1_bf, wfc2, fc2_b, x2, (float*)d_out, nullptr, (int)NX, kD, kF, 1.f, 0);
}
